// SimpleZoneODE_42880953483898
// MI455X (gfx1250) — matrix operand order repaired and checked by reading, not compiled, not run on hardware
//
#include <hip/hip_runtime.h>

typedef __attribute__((ext_vector_type(16))) _Float16 v16h;
typedef __attribute__((ext_vector_type(8)))  float    v8f;

#define N_ZONES_C 100000   // == 6250 * 16, tiles divide exactly (no tail handling needed)
#define HID_C     32
#define K1_C      56       // HID + PA + 16
#define N1_C      64       // 2*HID
#define LDS_STRIDE 68      // 64 + 4 pad -> rows 4 banks apart, conflict-free at float4 granularity

// ws layout:
//   bytes [0, 15*32*32)          : v16h ops[15][32]   (op x lane, 32B per lane)
//     ops 0..3  = B layer1 kstep0 (Wd1), nt = op
//     ops 4..7  = B layer1 kstep1 (Wd1, K-padded), nt = op-4
//     ops 8..9  = B layer2 kstep0 (Wd2), nt = op-8
//     ops 10..11= B layer2 kstep1 (Wd2), nt = op-10
//     ops 12..13= B layer3        (Wd3), nt = op-12
//     op  14    = A layer1 kstep1 (person_attrs + time-MLP + pad)
//   bytes [16384, 16384+32*8*4)  : float bias[32][8]  (lane-major: 4x bd1, 2x bd2, 2x bd3)
#define WS_BIAS_OFF 16384

__device__ __forceinline__ float relu_f(float x) { return x > 0.f ? x : 0.f; }

__device__ __forceinline__ float time_vec_at(int n, float t,
    const float* __restrict__ Wt1, const float* __restrict__ bt1,
    const float* __restrict__ Wt2, const float* __restrict__ bt2) {
  float acc = bt2[n];
#pragma unroll
  for (int j = 0; j < 16; ++j) {
    float hj = relu_f(t * Wt1[j] + bt1[j]);
    acc += hj * Wt2[j * 16 + n];
  }
  return acc;
}

// ---------- prep kernel: build all loop-invariant WMMA operands once ----------
// grid = 15 workgroups x 32 threads; block b builds operand b.
__global__ __launch_bounds__(32) void zone_ode_prep_kernel(
    const float* __restrict__ t_ptr,
    const float* __restrict__ person_attrs,
    const float* __restrict__ Wt1, const float* __restrict__ bt1,
    const float* __restrict__ Wt2, const float* __restrict__ bt2,
    const float* __restrict__ Wd1, const float* __restrict__ bd1,
    const float* __restrict__ Wd2, const float* __restrict__ bd2,
    const float* __restrict__ Wd3, const float* __restrict__ bd3,
    void* __restrict__ ws)
{
  const int op   = blockIdx.x;
  const int lane = threadIdx.x & 31;
  const int h    = lane >> 4;
  const int l16  = lane & 15;

  v16h val;
  if (op < 4) {                       // B layer1 kstep0: K = 8h+(i<8?i:i+8), col = nt*16+l16
    const int col = op * 16 + l16;
#pragma unroll
    for (int i = 0; i < 16; ++i)
      val[i] = (_Float16)Wd1[(8 * h + (i < 8 ? i : i + 8)) * N1_C + col];
  } else if (op < 8) {                // B layer1 kstep1: K = 32+8h+(i<8?i:i+8), pad K>=56 with 0
    const int col = (op - 4) * 16 + l16;
#pragma unroll
    for (int i = 0; i < 16; ++i) {
      int k = 32 + 8 * h + (i < 8 ? i : i + 8);
      float wv = Wd1[(k < K1_C ? k : 0) * N1_C + col];
      val[i] = (k < K1_C) ? (_Float16)wv : (_Float16)0.f;
    }
  } else if (op < 10) {               // B layer2 kstep0
    const int col = (op - 8) * 16 + l16;
#pragma unroll
    for (int i = 0; i < 16; ++i)
      val[i] = (_Float16)Wd2[(8 * h + (i < 8 ? i : i + 8)) * HID_C + col];
  } else if (op < 12) {               // B layer2 kstep1
    const int col = (op - 10) * 16 + l16;
#pragma unroll
    for (int i = 0; i < 16; ++i)
      val[i] = (_Float16)Wd2[(32 + 8 * h + (i < 8 ? i : i + 8)) * HID_C + col];
  } else if (op < 14) {               // B layer3
    const int col = (op - 12) * 16 + l16;
#pragma unroll
    for (int i = 0; i < 16; ++i)
      val[i] = (_Float16)Wd3[(8 * h + (i < 8 ? i : i + 8)) * HID_C + col];
  } else {                            // A layer1 kstep1: person_attrs + time_vec + pad
    const float t = t_ptr[0];
#pragma unroll
    for (int i = 0; i < 16; ++i) {
      int k = 32 + 8 * h + (i < 8 ? i : i + 8);
      float v;
      if (k < 40)        v = person_attrs[k - 32];
      else if (k < K1_C) v = time_vec_at(k - 40, t, Wt1, bt1, Wt2, bt2);
      else               v = 0.f;
      val[i] = (_Float16)v;
    }
    // this block also writes the per-lane bias table
    float* wsb = (float*)((char*)ws + WS_BIAS_OFF) + lane * 8;
#pragma unroll
    for (int nt = 0; nt < 4; ++nt) wsb[nt]     = bd1[nt * 16 + l16];
#pragma unroll
    for (int nt = 0; nt < 2; ++nt) wsb[4 + nt] = bd2[nt * 16 + l16];
#pragma unroll
    for (int nt = 0; nt < 2; ++nt) wsb[6 + nt] = bd3[nt * 16 + l16];
  }
  ((v16h*)ws)[op * 32 + lane] = val;
}

// ---------- main kernel: one 16-row zone tile per wave ----------
__global__ __launch_bounds__(256) void zone_ode_velocity_kernel(
    const float* __restrict__ zone_embedding,   // [N_ZONES, 32]
    const void*  __restrict__ ws,
    float* __restrict__ out)                    // [N_ZONES, 32]
{
  __shared__ float lds[8][16 * LDS_STRIDE];     // per-wave staging, ~34.8 KB

  const int lane = threadIdx.x & 31;
  const int wid  = threadIdx.x >> 5;
  const int h    = lane >> 4;
  const int l16  = lane & 15;

  float* __restrict__ myl = lds[wid];

  // wave-uniform tile id; N_ZONES is an exact multiple of 16 so no row guards anywhere
  const int n_tiles = N_ZONES_C / 16;           // 6250
  const int tile    = blockIdx.x * 8 + wid;
  if (tile >= n_tiles) return;                  // uniform per wave -> EXEC stays full
  const int tile_base = tile * 16;

  // ---- fetch all invariant operands: 15 x v16h (30x b128) + 2x float4 ----
  const v16h* __restrict__ wsv = (const v16h*)ws;
  v16h b1k0[4], b1k1[4], b2k0[2], b2k1[2], b3[2], a1;
#pragma unroll
  for (int nt = 0; nt < 4; ++nt) b1k0[nt] = wsv[(0 + nt) * 32 + lane];
#pragma unroll
  for (int nt = 0; nt < 4; ++nt) b1k1[nt] = wsv[(4 + nt) * 32 + lane];
#pragma unroll
  for (int nt = 0; nt < 2; ++nt) b2k0[nt] = wsv[(8 + nt) * 32 + lane];
#pragma unroll
  for (int nt = 0; nt < 2; ++nt) b2k1[nt] = wsv[(10 + nt) * 32 + lane];
#pragma unroll
  for (int nt = 0; nt < 2; ++nt) b3[nt]   = wsv[(12 + nt) * 32 + lane];
  a1 = wsv[14 * 32 + lane];

  const float4* __restrict__ wsb4 = (const float4*)((const char*)ws + WS_BIAS_OFF);
  const float4 bA = wsb4[lane * 2];       // bd1 cols nt=0..3
  const float4 bB = wsb4[lane * 2 + 1];   // bd2 nt=0..1, bd3 nt=0..1
  const float bias1[4] = { bA.x, bA.y, bA.z, bA.w };
  const float bias2[2] = { bB.x, bB.y };
  const float bias3[2] = { bB.z, bB.w };

  // ---- A operand, layer-1 kstep 0: zone_embedding columns (K = 0..31) ----
  v16h a0;
  {
    const float4* ze4 =
        reinterpret_cast<const float4*>(zone_embedding + (long)(tile_base + l16) * HID_C);
    float4 q0 = ze4[2 * h];         // K = 8h   .. 8h+3
    float4 q1 = ze4[2 * h + 1];     // K = 8h+4 .. 8h+7
    float4 q2 = ze4[4 + 2 * h];     // K = 16+8h   .. +3
    float4 q3 = ze4[5 + 2 * h];     // K = 16+8h+4 .. +7
    a0[0]  = (_Float16)q0.x; a0[1]  = (_Float16)q0.y; a0[2]  = (_Float16)q0.z; a0[3]  = (_Float16)q0.w;
    a0[4]  = (_Float16)q1.x; a0[5]  = (_Float16)q1.y; a0[6]  = (_Float16)q1.z; a0[7]  = (_Float16)q1.w;
    a0[8]  = (_Float16)q2.x; a0[9]  = (_Float16)q2.y; a0[10] = (_Float16)q2.z; a0[11] = (_Float16)q2.w;
    a0[12] = (_Float16)q3.x; a0[13] = (_Float16)q3.y; a0[14] = (_Float16)q3.z; a0[15] = (_Float16)q3.w;
  }

  // ---- Layer 1: combined[16x56] @ Wd1[56x64] ----
  v8f acc1[4];
#pragma unroll
  for (int nt = 0; nt < 4; ++nt) {
    v8f c = {};
    c = __builtin_amdgcn_wmma_f32_16x16x32_f16(false, a0, false, b1k0[nt], (short)0, c, false, false);
    c = __builtin_amdgcn_wmma_f32_16x16x32_f16(false, a1, false, b1k1[nt], (short)0, c, false, false);
    acc1[nt] = c;
  }

  // bias + relu, stage h1 [16x64] into per-wave LDS (D layout scatter)
#pragma unroll
  for (int nt = 0; nt < 4; ++nt) {
    const int col = nt * 16 + l16;
#pragma unroll
    for (int r = 0; r < 8; ++r) {
      int row = r + 8 * h;
      myl[row * LDS_STRIDE + col] = relu_f(acc1[nt][r] + bias1[nt]);
    }
  }

  // ---- Layer 2: h1[16x64] @ Wd2[64x32] ----
  v16h a2_0, a2_1;
  {
    const float* lr = myl + l16 * LDS_STRIDE;   // A row = lane & 15
#pragma unroll
    for (int i = 0; i < 16; ++i) {
      int kA = 8 * h + (i < 8 ? i : i + 8);
      a2_0[i] = (_Float16)lr[kA];        // kstep 0: K 0..31
      a2_1[i] = (_Float16)lr[32 + kA];   // kstep 1: K 32..63
    }
  }

  v8f acc2[2];
#pragma unroll
  for (int nt = 0; nt < 2; ++nt) {
    v8f c = {};
    c = __builtin_amdgcn_wmma_f32_16x16x32_f16(false, a2_0, false, b2k0[nt], (short)0, c, false, false);
    c = __builtin_amdgcn_wmma_f32_16x16x32_f16(false, a2_1, false, b2k1[nt], (short)0, c, false, false);
    acc2[nt] = c;
  }

  // bias + relu, stage h2 [16x32] into the same per-wave LDS region
#pragma unroll
  for (int nt = 0; nt < 2; ++nt) {
    const int col = nt * 16 + l16;
#pragma unroll
    for (int r = 0; r < 8; ++r) {
      int row = r + 8 * h;
      myl[row * LDS_STRIDE + col] = relu_f(acc2[nt][r] + bias2[nt]);
    }
  }

  // ---- Layer 3: h2[16x32] @ Wd3[32x32], + bd3, store ----
  v16h a3;
  {
    const float* lr = myl + l16 * LDS_STRIDE;
#pragma unroll
    for (int i = 0; i < 16; ++i)
      a3[i] = (_Float16)lr[8 * h + (i < 8 ? i : i + 8)];
  }

#pragma unroll
  for (int nt = 0; nt < 2; ++nt) {
    v8f c = {};
    c = __builtin_amdgcn_wmma_f32_16x16x32_f16(false, a3, false, b3[nt], (short)0, c, false, false);
    const int col = nt * 16 + l16;
    // one base address per nt; rows are a fixed 128B stride -> immediate offsets
    float* __restrict__ orow = out + (long)(tile_base + 8 * h) * HID_C + col;
#pragma unroll
    for (int r = 0; r < 8; ++r)
      orow[r * HID_C] = c[r] + bias3[nt];
  }
}

extern "C" void kernel_launch(void* const* d_in, const int* in_sizes, int n_in,
                              void* d_out, int out_size, void* d_ws, size_t ws_size,
                              hipStream_t stream) {
  // setup_inputs() order:
  // 0:t 1:zone_embedding 2:zone_features 3:person_attrs 4:edge_index
  // 5:W1 6:b1 7:W2 8:b2 9:Wt1 10:bt1 11:Wt2 12:bt2
  // 13:Wd1 14:bd1 15:Wd2 16:bd2 17:Wd3 18:bd3
  const float* t_ptr = (const float*)d_in[0];
  const float* ze    = (const float*)d_in[1];
  const float* pa    = (const float*)d_in[3];
  const float* Wt1   = (const float*)d_in[9];
  const float* bt1   = (const float*)d_in[10];
  const float* Wt2   = (const float*)d_in[11];
  const float* bt2   = (const float*)d_in[12];
  const float* Wd1   = (const float*)d_in[13];
  const float* bd1   = (const float*)d_in[14];
  const float* Wd2   = (const float*)d_in[15];
  const float* bd2   = (const float*)d_in[16];
  const float* Wd3   = (const float*)d_in[17];
  const float* bd3   = (const float*)d_in[18];
  float* out = (float*)d_out;

  // 1) build invariant WMMA operands into d_ws (15 single-wave blocks)
  zone_ode_prep_kernel<<<15, 32, 0, stream>>>(
      t_ptr, pa, Wt1, bt1, Wt2, bt2, Wd1, bd1, Wd2, bd2, Wd3, bd3, d_ws);

  // 2) main pass: 782 blocks x 8 waves -> one 16-row tile per wave (6250 tiles)
  const int n_tiles = N_ZONES_C / 16;                 // 6250
  const int blocks  = (n_tiles + 7) / 8;              // 782
  zone_ode_velocity_kernel<<<blocks, 256, 0, stream>>>(ze, d_ws, out);
}
